// Mixtral_13838384627730
// MI455X (gfx1250) — compile-verified, matrix operand order checked by reading, not run on hardware
//
#include <hip/hip_runtime.h>

// ---------------- problem constants ----------------
constexpr int T_TOK = 2048;   // B*S tokens
constexpr int DDIM  = 2048;   // model dim
constexpr int NEXP  = 8;      // experts
constexpr int FDIM  = 5632;   // mlp inner dim
constexpr int BM    = 128;    // M tile (token slots)
constexpr int BK    = 32;     // K tile (bf16 wmma K)
constexpr int LDSP  = 40;     // padded LDS row stride in ushorts (80B, 16B aligned)

// ---------------- workspace layout (bytes) ----------------
constexpr size_t OFF_XB   = 0;                                     // bf16 x  [T, D]
constexpr size_t SZ_XB    = (size_t)T_TOK * DDIM * 2;              // 8 MB
constexpr size_t OFF_COMB = OFF_XB + SZ_XB;                        // f32 comb [T, 8]
constexpr size_t OFF_CNT  = OFF_COMB + (size_t)T_TOK * 8 * 4;      // int cnt[8] (+pad)
constexpr size_t OFF_ROWS = OFF_CNT + 256;                         // int rows [E, T]
constexpr size_t OFF_H    = OFF_ROWS + (size_t)NEXP * T_TOK * 4;   // bf16 H [T, F]

// ---------------- vector types ----------------
typedef __attribute__((ext_vector_type(16))) __bf16 v16bf;
typedef __attribute__((ext_vector_type(8)))  float  v8f;

// ---------------- helpers ----------------
__device__ inline unsigned short f2bf(float f) {
  unsigned u = __float_as_uint(f);
  u += 0x7FFFu + ((u >> 16) & 1u);          // round-to-nearest-even
  return (unsigned short)(u >> 16);
}

__device__ inline uint2 pack4(float4 v) {
  uint2 r;
  r.x = (unsigned)f2bf(v.x) | ((unsigned)f2bf(v.y) << 16);
  r.y = (unsigned)f2bf(v.z) | ((unsigned)f2bf(v.w) << 16);
  return r;
}

__device__ inline v8f vzero8() {
  v8f z = {0.f, 0.f, 0.f, 0.f, 0.f, 0.f, 0.f, 0.f};
  return z;
}

// Low 32 bits of a flat pointer into LDS == wave-relative LDS byte offset
// (flat LDS aperture address = {SHARED_BASE, offset32}).
__device__ inline unsigned lds_off(const void* p) {
  return (unsigned)(size_t)p;
}

// CDNA5 async copy global -> LDS, 16B per lane, tracked on ASYNCcnt.
__device__ inline void async_ld_b128(unsigned ldsoff, const void* gaddr) {
  asm volatile("global_load_async_to_lds_b128 %0, %1, off"
               : : "v"(ldsoff), "v"(gaddr) : "memory");
}

__device__ inline void wait_async0() {
  asm volatile("s_wait_asynccnt 0x0" : : : "memory");
}

// gfx1250 16-bit A/B fragment: lane (lr = lane%16, lh = lane/16) holds
// K = {8*lh .. 8*lh+7} and {8*lh+16 .. 8*lh+23}  -> two b128 LDS loads.
__device__ inline v16bf ldfrag(const unsigned short* rowbase, int lh) {
  v16bf f;
  uint4* p = reinterpret_cast<uint4*>(&f);
  p[0] = *reinterpret_cast<const uint4*>(rowbase + 8 * lh);
  p[1] = *reinterpret_cast<const uint4*>(rowbase + 8 * lh + 16);
  return f;
}

__device__ inline v8f bfwmma(v16bf a, v16bf b, v8f c) {
  return __builtin_amdgcn_wmma_f32_16x16x32_bf16(
      /*neg_a=*/false, a, /*neg_b=*/false, b,
      /*c_mod=*/(short)0, c, /*reuse_a=*/false, /*reuse_b=*/false);
}

// ---------------- kernel 0: zero out + counters ----------------
__global__ __launch_bounds__(256)
void moe_init(float4* __restrict__ out4, int* __restrict__ cnt) {
  int i = blockIdx.x * 256 + threadIdx.x;         // 4096*256 == T*D/4
  out4[i] = make_float4(0.f, 0.f, 0.f, 0.f);
  if (i < NEXP) cnt[i] = 0;
}

// ---------------- kernel 1: x fp32 -> bf16 ----------------
__global__ __launch_bounds__(256)
void moe_cvt(const float4* __restrict__ xin4, uint2* __restrict__ xb4) {
  int i = blockIdx.x * 256 + threadIdx.x;         // T*D/4 elements
  xb4[i] = pack4(xin4[i]);
}

// ---------------- kernel 2: gate (top-2 softmax + routing lists) ----------------
__global__ __launch_bounds__(256)
void moe_gate(const float* __restrict__ x, const float* __restrict__ gw,
              float* __restrict__ comb, int* __restrict__ cnt,
              int* __restrict__ rowsb) {
  const int t = blockIdx.x;
  const int tid = threadIdx.x;
  __shared__ float red[8][8];
  __shared__ float lg[8];

  float acc[8];
#pragma unroll
  for (int e = 0; e < 8; ++e) acc[e] = 0.f;

  const float* xr = x + (size_t)t * DDIM;
  for (int d = tid; d < DDIM; d += 256) {
    float xv = xr[d];
#pragma unroll
    for (int e = 0; e < 8; ++e) acc[e] += xv * gw[e * DDIM + d];
  }
#pragma unroll
  for (int e = 0; e < 8; ++e) {
    for (int off = 16; off > 0; off >>= 1)
      acc[e] += __shfl_down(acc[e], off, 32);
  }
  const int wid = tid >> 5, lane = tid & 31;
  if (lane == 0) {
#pragma unroll
    for (int e = 0; e < 8; ++e) red[wid][e] = acc[e];
  }
  __syncthreads();
  if (tid < 8) {
    float s = 0.f;
#pragma unroll
    for (int w = 0; w < 8; ++w) s += red[w][tid];
    lg[tid] = s;
  }
  __syncthreads();
  if (tid == 0) {
    int i1 = 0;
#pragma unroll
    for (int e = 1; e < 8; ++e) if (lg[e] > lg[i1]) i1 = e;
    int i2 = (i1 == 0) ? 1 : 0;
#pragma unroll
    for (int e = 0; e < 8; ++e) if (e != i1 && lg[e] > lg[i2]) i2 = e;
    float p1 = 1.f / (1.f + __expf(lg[i2] - lg[i1]));   // softmax over top-2
    float p2 = 1.f - p1;
#pragma unroll
    for (int e = 0; e < 8; ++e)
      comb[t * 8 + e] = (e == i1) ? p1 : ((e == i2) ? p2 : 0.f);
    int pos1 = atomicAdd(&cnt[i1], 1); rowsb[i1 * T_TOK + pos1] = t;
    int pos2 = atomicAdd(&cnt[i2], 1); rowsb[i2 * T_TOK + pos2] = t;
  }
}

// ---------------- kernel 3: fused W1/W3 + SwiGLU -> H (bf16) ----------------
// C tile 128(M slots) x 64(N of F). 8 waves as 4(M)x2(N); per wave 2x2 frags
// for each of the two matmuls (X W1^T and X W3^T).
__global__ __launch_bounds__(256)
void moe_gemm1(int e, const unsigned short* __restrict__ xb,
               const float* __restrict__ w1, const float* __restrict__ w3,
               const int* __restrict__ rowsb, const int* __restrict__ cnt,
               unsigned short* __restrict__ Hbuf) {
  const int cnt_e = cnt[e];
  const int m0 = blockIdx.y * BM;
  if (m0 >= cnt_e) return;                      // routed-token tile skip
  const int n0 = blockIdx.x * 64;

  __shared__ unsigned short At[BM * LDSP];
  __shared__ unsigned short Bt1[64 * LDSP];
  __shared__ unsigned short Bt3[64 * LDSP];
  __shared__ int rowTok[BM];

  const int tid = threadIdx.x;
  if (tid < BM) {
    int s = m0 + tid;
    rowTok[tid] = rowsb[(size_t)e * T_TOK + min(s, cnt_e - 1)];
  }
  __syncthreads();

  const float* W1e = w1 + (size_t)e * FDIM * DDIM;
  const float* W3e = w3 + (size_t)e * FDIM * DDIM;

  // ---- per-thread chunk assignment, hoisted out of the K loop ----
  // A tile: 128 rows x 64B, 512 x b128 chunks, 2 per thread (async -> LDS).
  const int ar = tid >> 2, ap = tid & 3;
  const unsigned short* agb0 = xb + (size_t)rowTok[ar] * DDIM + ap * 8;
  const unsigned short* agb1 = xb + (size_t)rowTok[ar + 64] * DDIM + ap * 8;
  const unsigned aoff0 = lds_off(&At[ar * LDSP + ap * 8]);
  const unsigned aoff1 = lds_off(&At[(ar + 64) * LDSP + ap * 8]);
  // B tiles: 64 rows x 128B fp32, 512 x float4 chunks per matrix, 2 per thread.
  const int br = tid >> 3, bp = tid & 7;
  const float* b1g0 = W1e + (size_t)(n0 + br) * DDIM + bp * 4;
  const float* b1g1 = W1e + (size_t)(n0 + br + 32) * DDIM + bp * 4;
  const float* b3g0 = W3e + (size_t)(n0 + br) * DDIM + bp * 4;
  const float* b3g1 = W3e + (size_t)(n0 + br + 32) * DDIM + bp * 4;
  unsigned short* bs0 = &Bt1[br * LDSP + bp * 4];
  unsigned short* bs1 = &Bt1[(br + 32) * LDSP + bp * 4];
  unsigned short* bs2 = &Bt3[br * LDSP + bp * 4];
  unsigned short* bs3 = &Bt3[(br + 32) * LDSP + bp * 4];

  const int wid = tid >> 5, lane = tid & 31, lr = lane & 15, lh = lane >> 4;
  const int wm = wid >> 1, wn = wid & 1;

  v8f accG[2][2], accU[2][2];
#pragma unroll
  for (int i = 0; i < 2; ++i)
#pragma unroll
    for (int j = 0; j < 2; ++j) { accG[i][j] = vzero8(); accU[i][j] = vzero8(); }

  for (int kb = 0; kb < DDIM; kb += BK) {
    // A tile: async copy global(bf16) -> LDS, no VGPR round trip (ASYNCcnt)
    async_ld_b128(aoff0, agb0 + kb);
    async_ld_b128(aoff1, agb1 + kb);
    // B tiles: stream fp32 weights, convert to bf16 in-register (DScnt)
    *reinterpret_cast<uint2*>(bs0) = pack4(*reinterpret_cast<const float4*>(b1g0 + kb));
    *reinterpret_cast<uint2*>(bs1) = pack4(*reinterpret_cast<const float4*>(b1g1 + kb));
    *reinterpret_cast<uint2*>(bs2) = pack4(*reinterpret_cast<const float4*>(b3g0 + kb));
    *reinterpret_cast<uint2*>(bs3) = pack4(*reinterpret_cast<const float4*>(b3g1 + kb));
    if (kb + BK < DDIM) {                        // global_prefetch_b8 next K slab
      __builtin_prefetch(b1g0 + kb + BK, 0, 1);
      __builtin_prefetch(b3g0 + kb + BK, 0, 1);
    }
    wait_async0();
    __syncthreads();

    v16bf a0 = ldfrag(&At[(wm * 32 + 0 + lr) * LDSP], lh);
    v16bf a1 = ldfrag(&At[(wm * 32 + 16 + lr) * LDSP], lh);
#pragma unroll
    for (int ni = 0; ni < 2; ++ni) {
      v16bf b1 = ldfrag(&Bt1[(wn * 32 + ni * 16 + lr) * LDSP], lh);
      v16bf b3 = ldfrag(&Bt3[(wn * 32 + ni * 16 + lr) * LDSP], lh);
      accG[0][ni] = bfwmma(a0, b1, accG[0][ni]);
      accG[1][ni] = bfwmma(a1, b1, accG[1][ni]);
      accU[0][ni] = bfwmma(a0, b3, accU[0][ni]);
      accU[1][ni] = bfwmma(a1, b3, accU[1][ni]);
    }
    __syncthreads();
  }

  // epilogue: H = silu(G) * U -> bf16 (compact slot rows)
#pragma unroll
  for (int mi = 0; mi < 2; ++mi)
#pragma unroll
    for (int ni = 0; ni < 2; ++ni)
#pragma unroll
      for (int r = 0; r < 8; ++r) {
        int slot = m0 + wm * 32 + mi * 16 + lh * 8 + r;
        if (slot < cnt_e) {
          int col = n0 + wn * 32 + ni * 16 + lr;
          float g = accG[mi][ni][r];
          float u = accU[mi][ni][r];
          float h = (g / (1.f + __expf(-g))) * u;
          Hbuf[(size_t)slot * FDIM + col] = f2bf(h);
        }
      }
}

// ---------------- kernel 4: H W2^T, scale by comb, accumulate out ----------------
// C tile 128(M slots) x 128(N of D). 8 waves as 4(M)x2(N); per wave 2x4 frags.
__global__ __launch_bounds__(256)
void moe_gemm2(int e, const unsigned short* __restrict__ Hbuf,
               const float* __restrict__ w2,
               const int* __restrict__ rowsb, const int* __restrict__ cnt,
               const float* __restrict__ comb, float* __restrict__ out) {
  const int cnt_e = cnt[e];
  const int m0 = blockIdx.y * BM;
  if (m0 >= cnt_e) return;
  const int n0 = blockIdx.x * 128;

  __shared__ unsigned short At[BM * LDSP];
  __shared__ unsigned short Bt[128 * LDSP];

  const int tid = threadIdx.x;
  const float* W2e = w2 + (size_t)e * DDIM * FDIM;

  // A tile chunks (compact bf16 H rows): 2 async b128 per thread.
  const int ar = tid >> 2, ap = tid & 3;
  const unsigned short* agb0 = Hbuf + (size_t)(m0 + ar) * FDIM + ap * 8;
  const unsigned short* agb1 = Hbuf + (size_t)(m0 + ar + 64) * FDIM + ap * 8;
  const unsigned aoff0 = lds_off(&At[ar * LDSP + ap * 8]);
  const unsigned aoff1 = lds_off(&At[(ar + 64) * LDSP + ap * 8]);
  // B tile chunks: 128 rows x 8 float4 = 1024, 4 per thread.
  const int br = tid >> 3, bp = tid & 7;
  const float* bg0 = W2e + (size_t)(n0 + br) * FDIM + bp * 4;
  const float* bg1 = W2e + (size_t)(n0 + br + 32) * FDIM + bp * 4;
  const float* bg2 = W2e + (size_t)(n0 + br + 64) * FDIM + bp * 4;
  const float* bg3 = W2e + (size_t)(n0 + br + 96) * FDIM + bp * 4;
  unsigned short* bs0 = &Bt[br * LDSP + bp * 4];
  unsigned short* bs1 = &Bt[(br + 32) * LDSP + bp * 4];
  unsigned short* bs2 = &Bt[(br + 64) * LDSP + bp * 4];
  unsigned short* bs3 = &Bt[(br + 96) * LDSP + bp * 4];

  const int wid = tid >> 5, lane = tid & 31, lr = lane & 15, lh = lane >> 4;
  const int wm = wid >> 1, wn = wid & 1;

  v8f acc[2][4];
#pragma unroll
  for (int i = 0; i < 2; ++i)
#pragma unroll
    for (int j = 0; j < 4; ++j) acc[i][j] = vzero8();

  for (int kb = 0; kb < FDIM; kb += BK) {
    async_ld_b128(aoff0, agb0 + kb);
    async_ld_b128(aoff1, agb1 + kb);
    *reinterpret_cast<uint2*>(bs0) = pack4(*reinterpret_cast<const float4*>(bg0 + kb));
    *reinterpret_cast<uint2*>(bs1) = pack4(*reinterpret_cast<const float4*>(bg1 + kb));
    *reinterpret_cast<uint2*>(bs2) = pack4(*reinterpret_cast<const float4*>(bg2 + kb));
    *reinterpret_cast<uint2*>(bs3) = pack4(*reinterpret_cast<const float4*>(bg3 + kb));
    if (kb + BK < FDIM) {
      __builtin_prefetch(bg0 + kb + BK, 0, 1);
      __builtin_prefetch(bg2 + kb + BK, 0, 1);
    }
    wait_async0();
    __syncthreads();

    v16bf a0 = ldfrag(&At[(wm * 32 + 0 + lr) * LDSP], lh);
    v16bf a1 = ldfrag(&At[(wm * 32 + 16 + lr) * LDSP], lh);
#pragma unroll
    for (int ni = 0; ni < 4; ++ni) {
      v16bf b = ldfrag(&Bt[(wn * 64 + ni * 16 + lr) * LDSP], lh);
      acc[0][ni] = bfwmma(a0, b, acc[0][ni]);
      acc[1][ni] = bfwmma(a1, b, acc[1][ni]);
    }
    __syncthreads();
  }

  // epilogue: out[tok] += comb[tok,e] * y   (unique per expert launch -> plain RMW)
  int   tok[2][8];
  float cw[2][8];
  bool  ok[2][8];
#pragma unroll
  for (int mi = 0; mi < 2; ++mi)
#pragma unroll
    for (int r = 0; r < 8; ++r) {
      int slot = m0 + wm * 32 + mi * 16 + lh * 8 + r;
      ok[mi][r] = (slot < cnt_e);
      int s = min(slot, cnt_e - 1);
      int tk = rowsb[(size_t)e * T_TOK + s];
      tok[mi][r] = tk;
      cw[mi][r] = comb[tk * 8 + e];
    }
#pragma unroll
  for (int mi = 0; mi < 2; ++mi)
#pragma unroll
    for (int ni = 0; ni < 4; ++ni)
#pragma unroll
      for (int r = 0; r < 8; ++r) {
        if (ok[mi][r]) {
          int col = n0 + wn * 64 + ni * 16 + lr;
          float* p = out + (size_t)tok[mi][r] * DDIM + col;
          *p += cw[mi][r] * acc[mi][ni][r];
        }
      }
}

// ---------------- host ----------------
extern "C" void kernel_launch(void* const* d_in, const int* in_sizes, int n_in,
                              void* d_out, int out_size, void* d_ws, size_t ws_size,
                              hipStream_t stream) {
  const float* x  = (const float*)d_in[0];   // stm  [1,2048,16,128] == [T, D]
  const float* gw = (const float*)d_in[1];   // gate_w [E, D]
  const float* w1 = (const float*)d_in[2];   // [E, F, D]
  const float* w2 = (const float*)d_in[3];   // [E, D, F]
  const float* w3 = (const float*)d_in[4];   // [E, F, D]
  float* out = (float*)d_out;

  char* ws = (char*)d_ws;
  unsigned short* xb   = (unsigned short*)(ws + OFF_XB);
  float*          comb = (float*)(ws + OFF_COMB);
  int*            cnt  = (int*)(ws + OFF_CNT);
  int*            rows = (int*)(ws + OFF_ROWS);
  unsigned short* Hbuf = (unsigned short*)(ws + OFF_H);

  const int n4blocks = (T_TOK * DDIM / 4) / 256;   // 4096
  moe_init<<<n4blocks, 256, 0, stream>>>((float4*)out, cnt);
  moe_cvt<<<n4blocks, 256, 0, stream>>>((const float4*)x, (uint2*)xb);
  moe_gate<<<T_TOK, 256, 0, stream>>>(x, gw, comb, cnt, rows);

  for (int e = 0; e < NEXP; ++e) {
    moe_gemm1<<<dim3(FDIM / 64, T_TOK / BM), 256, 0, stream>>>(
        e, xb, w1, w3, rows, cnt, Hbuf);
    moe_gemm2<<<dim3(DDIM / 128, T_TOK / BM), 256, 0, stream>>>(
        e, Hbuf, w2, rows, cnt, comb, out);
  }
}
